// ConvAttention_63711544869554
// MI455X (gfx1250) — compile-verified, matrix operand order checked by reading, not run on hardware
//
#include <hip/hip_runtime.h>

// ---------------------------------------------------------------------------
// ConvAttention forward for MI455X (gfx1250, wave32, WMMA).
// Heavy GEMMs use v_wmma_f32_16x16x32_bf16 (bf16 operands, f32 accumulate)
// fed from LDS via async global->LDS copies (ASYNCcnt) with double buffering.
// Problem is memory-bound on the a1 output (400MB); compute is ~55 GFLOP.
// ---------------------------------------------------------------------------

typedef __attribute__((ext_vector_type(16))) __bf16 v16bf;
typedef __attribute__((ext_vector_type(8)))  __bf16 v8bf;
typedef __attribute__((ext_vector_type(8)))  float  v8f;
typedef __attribute__((ext_vector_type(4)))  float  v4f;
typedef __attribute__((ext_vector_type(4)))  int    v4i;

#define DMODEL 768
#define HEADS  6
#define HSIZE  64
#define DIM    384
#define KER    9
#define SEQ    2048
#define BATCH  4
#define ROWS   (BATCH * SEQ)     // 8192
#define BH     (BATCH * HEADS)   // 24
#define NQKV   (4 * DIM)         // 1536 : [q | k | v | cox]
#define NINF_C (-10000.0f)
#define PADROW 40                // LDS row pitch in bf16 (32 data + 8 pad = 80B)

// --- async global->LDS support (gfx1250) -----------------------------------
#if defined(__AMDGCN__) && __has_builtin(__builtin_amdgcn_global_load_async_to_lds_b128)
#define HAVE_ASYNC_LDS 1
#else
#define HAVE_ASYNC_LDS 0
#endif

#if HAVE_ASYNC_LDS
typedef __attribute__((address_space(1))) v4i* gptr_v4i;
typedef __attribute__((address_space(3))) v4i* lptr_v4i;
#define ASYNC_COPY16(gp, lp)                                                   \
  __builtin_amdgcn_global_load_async_to_lds_b128((gptr_v4i)(gp), (lptr_v4i)(lp), 0, 0)
#if __has_builtin(__builtin_amdgcn_s_wait_asynccnt)
#define ASYNC_WAIT() __builtin_amdgcn_s_wait_asynccnt(0)
#else
#define ASYNC_WAIT() asm volatile("s_wait_asynccnt 0" ::: "memory")
#endif
#else
#define ASYNC_COPY16(gp, lp) (*(v8bf*)(lp) = *(const v8bf*)(gp))
#define ASYNC_WAIT()
#endif

union FragBF { v16bf f; v8bf h[2]; };

// Fragment gather matching the CDNA5 16-bit A/B VGPR layout:
// lane (idx<16 -> half g=0, else g=1): elements 0..7  = K (k0+8g .. +7)
//                                      elements 8..15 = K (k0+16+8g .. +7)
__device__ __forceinline__ v16bf load_frag_bf16(const __bf16* __restrict__ base,
                                                int ld, int row, int kofs) {
  FragBF fr;
  const __bf16* p = base + (size_t)row * ld + kofs;
  fr.h[0] = *(const v8bf*)(p);
  fr.h[1] = *(const v8bf*)(p + 16);
  return fr.f;
}

// Same gather but from an LDS tile row (row pitch PADROW, k-span 32)
__device__ __forceinline__ v16bf frag_lds(const __bf16* row, int g) {
  FragBF fr;
  fr.h[0] = *(const v8bf*)(row + 8 * g);
  fr.h[1] = *(const v8bf*)(row + 16 + 8 * g);
  return fr.f;
}

__device__ __forceinline__ v8f wmma_bf16(v16bf a, v16bf b, v8f c) {
  return __builtin_amdgcn_wmma_f32_16x16x32_bf16(false, a, false, b,
                                                 (short)0, c, false, false);
}

// ---------------------------------------------------------------------------
// Prep kernels
// ---------------------------------------------------------------------------
__global__ void k_cvt_bf16(const float* __restrict__ in, __bf16* __restrict__ out, int n) {
  int i = blockIdx.x * blockDim.x + threadIdx.x;
  if (i < n) out[i] = (__bf16)in[i];
}

// Wt[n*K + k] = (bf16) W[k*N + n]   (W is KxN row-major)
__global__ void k_transpose_cvt(const float* __restrict__ W, __bf16* __restrict__ Wt,
                                int K, int N) {
  int i = blockIdx.x * blockDim.x + threadIdx.x;
  if (i >= K * N) return;
  int n = i / K, k = i - n * K;
  Wt[i] = (__bf16)W[(size_t)k * N + n];
}

__global__ void k_bias_concat(const float* __restrict__ bq, const float* __restrict__ bk,
                              const float* __restrict__ bv, const float* __restrict__ bco,
                              float* __restrict__ out) {
  int i = blockIdx.x * blockDim.x + threadIdx.x;
  if (i >= NQKV) return;
  int sel = i / DIM, c = i - sel * DIM;
  float v = (sel == 0) ? bq[c] : (sel == 1) ? bk[c] : (sel == 2) ? bv[c] : bco[c];
  out[i] = v;
}

// Depthwise SAME conv (K=9, pad 4/4) -> bf16 (feeds the ccx GEMM)
__global__ void k_dwconv(const float* __restrict__ x, const float* __restrict__ dw,
                         __bf16* __restrict__ y) {
  int i = blockIdx.x * blockDim.x + threadIdx.x;
  if (i >= ROWS * DMODEL) return;
  int c = i % DMODEL, row = i / DMODEL;
  int l = row & (SEQ - 1), b = row / SEQ;
  float s = 0.f;
#pragma unroll
  for (int t = 0; t < KER; ++t) {
    int ll = l + t - 4;
    if ((unsigned)ll < (unsigned)SEQ)
      s += x[((size_t)(b * SEQ + ll)) * DMODEL + c] * dw[t * DMODEL + c];
  }
  y[i] = (__bf16)s;
}

// Split q/k/v into per-head bf16 layouts. Qh,Kh: [BH][L][64]; Vt: [BH][64][L]
__global__ void k_split_heads(const float* __restrict__ qkvc,
                              __bf16* __restrict__ Qh, __bf16* __restrict__ Kh,
                              __bf16* __restrict__ Vt) {
  int i = blockIdx.x * blockDim.x + threadIdx.x;
  if (i >= BH * SEQ * HSIZE) return;
  int s  = i & (HSIZE - 1);
  int l  = (i >> 6) & (SEQ - 1);
  int bh = i >> 17;                       // / (SEQ*HSIZE)
  int b = bh / HEADS, h = bh - b * HEADS;
  size_t src = ((size_t)(b * SEQ + l)) * NQKV + h * HSIZE + s;
  Qh[i] = (__bf16)qkvc[src];
  Kh[i] = (__bf16)qkvc[src + DIM];
  Vt[(size_t)bh * (HSIZE * SEQ) + (size_t)s * SEQ + l] = (__bf16)qkvc[src + 2 * DIM];
}

// ---------------------------------------------------------------------------
// Generic bf16 WMMA GEMM with async-LDS double buffering:
//   C[M x N] = A[M x K](bf16) * Bt[N x K](bf16)^T + bias[N] (+ resid[M x N])
// 8 waves/WG (4 in M, 2 in N) -> WG tile 128 x 64, wave tile 32 x 32.
// Per K-step the WG stages A(128x32) and B(64x32) into LDS via
// GLOBAL_LOAD_ASYNC_TO_LDS_B128 (ASYNCcnt + barrier), overlapped with WMMA.
// grid = (N/64, M/128)
// ---------------------------------------------------------------------------
__global__ __launch_bounds__(256) void k_gemm_bf16(
    const __bf16* __restrict__ A, int lda,
    const __bf16* __restrict__ Bt, int ldb,
    const float* __restrict__ bias, const float* __restrict__ resid,
    float* __restrict__ C, int ldc, int K) {
  __shared__ __bf16 Abuf[2][128 * PADROW];   // 2 x 10240 B
  __shared__ __bf16 Bbuf[2][ 64 * PADROW];   // 2 x  5120 B
  const int tid  = threadIdx.x;
  const int lane = tid & 31;
  const int wave = tid >> 5;
  const int wm = wave >> 1, wn = wave & 1;
  const int ml = lane & 15, g = lane >> 4;
  const int m_blk = blockIdx.y * 128;
  const int n_blk = blockIdx.x * 64;
  const int m_base = wm * 32;                // within LDS tile
  const int n_base = wn * 32;

  // 16B-chunk copy plan: A = 512 chunks (tid, tid+256), B = 256 chunks (tid)
  const int am0 = tid >> 2,        ac0 = tid & 3;         // rows 0..63
  const int am1 = (tid + 256) >> 2, ac1 = tid & 3;        // rows 64..127
  const int bn  = tid >> 2,        bc  = tid & 3;
  auto issue_tile = [&](int buf, int k0) {
    ASYNC_COPY16(A  + (size_t)(m_blk + am0) * lda + k0 + ac0 * 8,
                 &Abuf[buf][am0 * PADROW + ac0 * 8]);
    ASYNC_COPY16(A  + (size_t)(m_blk + am1) * lda + k0 + ac1 * 8,
                 &Abuf[buf][am1 * PADROW + ac1 * 8]);
    ASYNC_COPY16(Bt + (size_t)(n_blk + bn) * ldb + k0 + bc * 8,
                 &Bbuf[buf][bn * PADROW + bc * 8]);
  };

  v8f acc[2][2] = {};
  issue_tile(0, 0);
  ASYNC_WAIT();
  __syncthreads();

  int cur = 0;
  for (int k0 = 0; k0 < K; k0 += 32) {
    if (k0 + 32 < K) issue_tile(cur ^ 1, k0 + 32);
    const __bf16* Ab = Abuf[cur];
    const __bf16* Bb = Bbuf[cur];
    v16bf a0 = frag_lds(&Ab[(m_base + ml) * PADROW], g);
    v16bf a1 = frag_lds(&Ab[(m_base + 16 + ml) * PADROW], g);
    v16bf b0 = frag_lds(&Bb[(n_base + ml) * PADROW], g);
    v16bf b1 = frag_lds(&Bb[(n_base + 16 + ml) * PADROW], g);
    acc[0][0] = wmma_bf16(a0, b0, acc[0][0]);
    acc[0][1] = wmma_bf16(a0, b1, acc[0][1]);
    acc[1][0] = wmma_bf16(a1, b0, acc[1][0]);
    acc[1][1] = wmma_bf16(a1, b1, acc[1][1]);
    ASYNC_WAIT();          // own async copies into buf^1 done
    __syncthreads();       // all waves done reading buf `cur`
    cur ^= 1;
  }

#pragma unroll
  for (int j = 0; j < 2; ++j) {
    int n = n_blk + n_base + j * 16 + ml;
    float bv = bias ? bias[n] : 0.f;
#pragma unroll
    for (int i = 0; i < 2; ++i) {
#pragma unroll
      for (int r = 0; r < 8; ++r) {
        int m = m_blk + m_base + i * 16 + g * 8 + r;  // C/D: lanes16-31 -> M+8
        size_t idx = (size_t)m * ldc + n;
        float val = acc[i][j][r] + bv;
        if (resid) val += resid[idx];
        C[idx] = val;
      }
    }
  }
}

// ---------------------------------------------------------------------------
// Attention scores: a1_raw[bh][q][k] = (Qh . Kh)/8 + NINF*mask[b][k]
// Store-bound (writes 400MB); operands are L2-resident -> direct global.
// grid = (2048/64, 2048/128, BH)
// ---------------------------------------------------------------------------
__global__ __launch_bounds__(256) void k_scores(
    const __bf16* __restrict__ Qh, const __bf16* __restrict__ Kh,
    const int* __restrict__ mask, float* __restrict__ a1) {
  const int lane = threadIdx.x & 31;
  const int wave = threadIdx.x >> 5;
  const int wm = wave >> 1, wn = wave & 1;
  const int ml = lane & 15, g = lane >> 4;
  const int m_base = blockIdx.y * 128 + wm * 32;
  const int n_base = blockIdx.x * 64 + wn * 32;
  const int bh = blockIdx.z;
  const __bf16* Aq = Qh + (size_t)bh * SEQ * HSIZE;
  const __bf16* Bk = Kh + (size_t)bh * SEQ * HSIZE;
  float* out = a1 + (size_t)bh * SEQ * SEQ;
  const int* mrow = mask + (size_t)(bh / HEADS) * SEQ;

  v8f acc[2][2] = {};
#pragma unroll
  for (int k0 = 0; k0 < HSIZE; k0 += 32) {
    int kofs = k0 + 8 * g;
    v16bf a0 = load_frag_bf16(Aq, HSIZE, m_base + ml,      kofs);
    v16bf a1f = load_frag_bf16(Aq, HSIZE, m_base + 16 + ml, kofs);
    v16bf b0 = load_frag_bf16(Bk, HSIZE, n_base + ml,      kofs);
    v16bf b1 = load_frag_bf16(Bk, HSIZE, n_base + 16 + ml, kofs);
    acc[0][0] = wmma_bf16(a0,  b0, acc[0][0]);
    acc[0][1] = wmma_bf16(a0,  b1, acc[0][1]);
    acc[1][0] = wmma_bf16(a1f, b0, acc[1][0]);
    acc[1][1] = wmma_bf16(a1f, b1, acc[1][1]);
  }
#pragma unroll
  for (int j = 0; j < 2; ++j) {
    int n = n_base + j * 16 + ml;
    float madd = NINF_C * (float)mrow[n];
#pragma unroll
    for (int i = 0; i < 2; ++i) {
#pragma unroll
      for (int r = 0; r < 8; ++r) {
        int m = m_base + i * 16 + g * 8 + r;
        out[(size_t)m * SEQ + n] = acc[i][j][r] * 0.125f + madd;
      }
    }
  }
}

// ---------------------------------------------------------------------------
// Row softmax in place over a1 (49152 rows of 2048)
// ---------------------------------------------------------------------------
__global__ __launch_bounds__(256) void k_softmax(float* __restrict__ a1) {
  __shared__ float red[256];
  const int tid = threadIdx.x;
  float* row = a1 + (size_t)blockIdx.x * SEQ;
  float v[8];
  float mx = -3.4e38f;
#pragma unroll
  for (int j = 0; j < 8; ++j) { v[j] = row[tid + j * 256]; mx = fmaxf(mx, v[j]); }
  red[tid] = mx; __syncthreads();
  for (int s = 128; s > 0; s >>= 1) {
    if (tid < s) red[tid] = fmaxf(red[tid], red[tid + s]);
    __syncthreads();
  }
  mx = red[0]; __syncthreads();
  float sum = 0.f;
#pragma unroll
  for (int j = 0; j < 8; ++j) { v[j] = __expf(v[j] - mx); sum += v[j]; }
  red[tid] = sum; __syncthreads();
  for (int s = 128; s > 0; s >>= 1) {
    if (tid < s) red[tid] += red[tid + s];
    __syncthreads();
  }
  float inv = 1.f / red[0];
#pragma unroll
  for (int j = 0; j < 8; ++j) row[tid + j * 256] = v[j] * inv;
}

// ---------------------------------------------------------------------------
// ctx = a1(f32 -> bf16 in-register) @ V : writes bf16 into cat[:, 0:384]
// a1 is streamed exactly once (read-bound) -> direct global.
// grid = (1, 2048/128, BH); wave n-split covers S=64
// ---------------------------------------------------------------------------
__global__ __launch_bounds__(256) void k_ctx(
    const float* __restrict__ a1, const __bf16* __restrict__ Vt,
    __bf16* __restrict__ catbf) {
  const int lane = threadIdx.x & 31;
  const int wave = threadIdx.x >> 5;
  const int wm = wave >> 1, wn = wave & 1;
  const int ml = lane & 15, g = lane >> 4;
  const int m_base = blockIdx.y * 128 + wm * 32;
  const int n_base = wn * 32;
  const int bh = blockIdx.z;
  const int b = bh / HEADS, h = bh - b * HEADS;
  const float*  Aa = a1 + (size_t)bh * SEQ * SEQ;
  const __bf16* Bv = Vt + (size_t)bh * HSIZE * SEQ;   // [64][2048]

  v8f acc[2][2] = {};
  for (int k0 = 0; k0 < SEQ; k0 += 32) {
    int kofs = k0 + 8 * g;
    FragBF af[2];
#pragma unroll
    for (int i = 0; i < 2; ++i) {
      const float* p = Aa + (size_t)(m_base + i * 16 + ml) * SEQ + kofs;
      v4f f0 = *(const v4f*)(p);
      v4f f1 = *(const v4f*)(p + 4);
      v4f f2 = *(const v4f*)(p + 16);
      v4f f3 = *(const v4f*)(p + 20);
#pragma unroll
      for (int t = 0; t < 4; ++t) {
        af[i].f[t]      = (__bf16)f0[t];
        af[i].f[4 + t]  = (__bf16)f1[t];
        af[i].f[8 + t]  = (__bf16)f2[t];
        af[i].f[12 + t] = (__bf16)f3[t];
      }
    }
    v16bf b0 = frag_lds(Bv + (size_t)(n_base + ml) * SEQ + k0, g);      // global,
    v16bf b1 = frag_lds(Bv + (size_t)(n_base + 16 + ml) * SEQ + k0, g); // same gather
    acc[0][0] = wmma_bf16(af[0].f, b0, acc[0][0]);
    acc[0][1] = wmma_bf16(af[0].f, b1, acc[0][1]);
    acc[1][0] = wmma_bf16(af[1].f, b0, acc[1][0]);
    acc[1][1] = wmma_bf16(af[1].f, b1, acc[1][1]);
  }
#pragma unroll
  for (int j = 0; j < 2; ++j) {
    int s = n_base + j * 16 + ml;               // 0..63
#pragma unroll
    for (int i = 0; i < 2; ++i) {
#pragma unroll
      for (int r = 0; r < 8; ++r) {
        int m = m_base + i * 16 + g * 8 + r;
        catbf[(size_t)(b * SEQ + m) * (2 * DIM) + h * HSIZE + s] =
            (__bf16)acc[i][j][r];
      }
    }
  }
}

// ---------------------------------------------------------------------------
// ck branch: t = q*ccx; ck_out = t @ ck_w + ck_b; per-head 9-way softmax -> lw
// one block (64 threads) per row
// ---------------------------------------------------------------------------
__global__ __launch_bounds__(64) void k_ck(
    const float* __restrict__ qkvc, const float* __restrict__ ccx,
    const float* __restrict__ ck_w, const float* __restrict__ ck_b,
    float* __restrict__ lw) {
  __shared__ float tbuf[DIM];
  __shared__ float cks[HEADS * KER];
  const int row = blockIdx.x, tid = threadIdx.x;
  for (int i = tid; i < DIM; i += 64)
    tbuf[i] = qkvc[(size_t)row * NQKV + i] * ccx[(size_t)row * DIM + i];
  __syncthreads();
  for (int j = tid; j < HEADS * KER; j += 64) {
    float s = ck_b[j];
    for (int c = 0; c < DIM; ++c) s += tbuf[c] * ck_w[c * (HEADS * KER) + j];
    cks[j] = s;
  }
  __syncthreads();
  if (tid < HEADS) {
    float mx = -3.4e38f;
#pragma unroll
    for (int kk = 0; kk < KER; ++kk) mx = fmaxf(mx, cks[tid * KER + kk]);
    float e[KER], sum = 0.f;
#pragma unroll
    for (int kk = 0; kk < KER; ++kk) { e[kk] = __expf(cks[tid * KER + kk] - mx); sum += e[kk]; }
    float inv = 1.f / sum;
#pragma unroll
    for (int kk = 0; kk < KER; ++kk)
      lw[(size_t)row * (HEADS * KER) + tid * KER + kk] = e[kk] * inv;
  }
}

// local[b,l,h,s] = sum_k lw[b,l,h,k] * cox[b, l+k-4, h*64+s] -> cat[:, 384:768]
__global__ void k_local(const float* __restrict__ lw, const float* __restrict__ qkvc,
                        __bf16* __restrict__ catbf) {
  int i = blockIdx.x * blockDim.x + threadIdx.x;
  if (i >= ROWS * DIM) return;
  int d = i % DIM, row = i / DIM;
  int h = d >> 6;
  int l = row & (SEQ - 1), b = row >> 11;
  const float* lwr = lw + (size_t)row * (HEADS * KER) + h * KER;
  float s = 0.f;
#pragma unroll
  for (int t = 0; t < KER; ++t) {
    int ll = l + t - 4;
    if ((unsigned)ll < (unsigned)SEQ)
      s += lwr[t] * qkvc[((size_t)(b * SEQ + ll)) * NQKV + 3 * DIM + d];
  }
  catbf[(size_t)row * (2 * DIM) + DIM + d] = (__bf16)s;
}

// ---------------------------------------------------------------------------
// LayerNorm: out = (h - mu) * rsqrt(var + eps) * gamma + beta ; one block/row
// ---------------------------------------------------------------------------
__global__ __launch_bounds__(256) void k_layernorm(
    const float* __restrict__ h, const float* __restrict__ gamma,
    const float* __restrict__ beta, float* __restrict__ out) {
  __shared__ float red[256];
  const int tid = threadIdx.x;
  const float* row = h + (size_t)blockIdx.x * DMODEL;
  float v[3];
  float sum = 0.f;
#pragma unroll
  for (int j = 0; j < 3; ++j) { v[j] = row[tid + j * 256]; sum += v[j]; }
  red[tid] = sum; __syncthreads();
  for (int s = 128; s > 0; s >>= 1) { if (tid < s) red[tid] += red[tid + s]; __syncthreads(); }
  float mu = red[0] * (1.0f / DMODEL); __syncthreads();
  float sq = 0.f;
#pragma unroll
  for (int j = 0; j < 3; ++j) { float d = v[j] - mu; sq += d * d; }
  red[tid] = sq; __syncthreads();
  for (int s = 128; s > 0; s >>= 1) { if (tid < s) red[tid] += red[tid + s]; __syncthreads(); }
  float rstd = rsqrtf(red[0] * (1.0f / DMODEL) + 1e-6f);
  float* orow = out + (size_t)blockIdx.x * DMODEL;
#pragma unroll
  for (int j = 0; j < 3; ++j) {
    int c = tid + j * 256;
    orow[c] = (v[j] - mu) * rstd * gamma[c] + beta[c];
  }
}

// ---------------------------------------------------------------------------
// Host launch
// ---------------------------------------------------------------------------
extern "C" void kernel_launch(void* const* d_in, const int* in_sizes, int n_in,
                              void* d_out, int out_size, void* d_ws, size_t ws_size,
                              hipStream_t stream) {
  const float* x       = (const float*)d_in[0];
  const int*   mask    = (const int*)  d_in[1];
  const float* wq_w    = (const float*)d_in[2];
  const float* wq_b    = (const float*)d_in[3];
  const float* wk_w    = (const float*)d_in[4];
  const float* wk_b    = (const float*)d_in[5];
  const float* wv_w    = (const float*)d_in[6];
  const float* wv_b    = (const float*)d_in[7];
  const float* dense_w = (const float*)d_in[8];
  const float* dense_b = (const float*)d_in[9];
  const float* cc_dw   = (const float*)d_in[10];
  const float* cc_pw   = (const float*)d_in[11];
  const float* cc_b    = (const float*)d_in[12];
  const float* ck_w    = (const float*)d_in[13];
  const float* ck_b    = (const float*)d_in[14];
  const float* co_w    = (const float*)d_in[15];
  const float* co_b    = (const float*)d_in[16];
  const float* gamma   = (const float*)d_in[17];
  const float* beta    = (const float*)d_in[18];

  float* out = (float*)d_out;                          // [B,L,768]
  float* a1  = (float*)d_out + (size_t)ROWS * DMODEL;  // [B,H,L,L]

  // workspace bump allocator (256B aligned)
  size_t off = 0;
  auto alloc = [&](size_t bytes) -> void* {
    off = (off + 255) & ~(size_t)255;
    void* p = (char*)d_ws + off;
    off += bytes;
    return p;
  };
  __bf16* Xbf    = (__bf16*)alloc((size_t)ROWS * DMODEL * 2);
  __bf16* Ybf    = (__bf16*)alloc((size_t)ROWS * DMODEL * 2);
  __bf16* WtAll  = (__bf16*)alloc((size_t)NQKV * DMODEL * 2);
  __bf16* CcpwT  = (__bf16*)alloc((size_t)DIM * DMODEL * 2);
  __bf16* DenseT = (__bf16*)alloc((size_t)DMODEL * DMODEL * 2);
  float*  biasAll= (float*) alloc((size_t)NQKV * 4);
  float*  qkvc   = (float*) alloc((size_t)ROWS * NQKV * 4);
  float*  ccx    = (float*) alloc((size_t)ROWS * DIM * 4);
  __bf16* Qh     = (__bf16*)alloc((size_t)BH * SEQ * HSIZE * 2);
  __bf16* Kh     = (__bf16*)alloc((size_t)BH * SEQ * HSIZE * 2);
  __bf16* Vt     = (__bf16*)alloc((size_t)BH * SEQ * HSIZE * 2);
  float*  lw     = (float*) alloc((size_t)ROWS * HEADS * KER * 4);
  __bf16* catbf  = (__bf16*)alloc((size_t)ROWS * 2 * DIM * 2);
  float*  hbuf   = (float*) alloc((size_t)ROWS * DMODEL * 4);

  const int T = 256;
  // --- prep: bf16 conversions / transposes ---
  k_cvt_bf16<<<(ROWS * DMODEL + T - 1) / T, T, 0, stream>>>(x, Xbf, ROWS * DMODEL);
  int wgrid = (DMODEL * DIM + T - 1) / T;
  k_transpose_cvt<<<wgrid, T, 0, stream>>>(wq_w, WtAll + (size_t)0 * DIM * DMODEL, DMODEL, DIM);
  k_transpose_cvt<<<wgrid, T, 0, stream>>>(wk_w, WtAll + (size_t)1 * DIM * DMODEL, DMODEL, DIM);
  k_transpose_cvt<<<wgrid, T, 0, stream>>>(wv_w, WtAll + (size_t)2 * DIM * DMODEL, DMODEL, DIM);
  k_transpose_cvt<<<wgrid, T, 0, stream>>>(co_w, WtAll + (size_t)3 * DIM * DMODEL, DMODEL, DIM);
  k_transpose_cvt<<<wgrid, T, 0, stream>>>(cc_pw, CcpwT, DMODEL, DIM);
  k_transpose_cvt<<<(DMODEL * DMODEL + T - 1) / T, T, 0, stream>>>(dense_w, DenseT, DMODEL, DMODEL);
  k_bias_concat<<<(NQKV + T - 1) / T, T, 0, stream>>>(wq_b, wk_b, wv_b, co_b, biasAll);
  k_dwconv<<<(ROWS * DMODEL + T - 1) / T, T, 0, stream>>>(x, cc_dw, Ybf);

  // --- projections: [q|k|v|cox] = x @ W  (M=8192, N=1536, K=768) ---
  k_gemm_bf16<<<dim3(NQKV / 64, ROWS / 128), T, 0, stream>>>(
      Xbf, DMODEL, WtAll, DMODEL, biasAll, nullptr, qkvc, NQKV, DMODEL);
  // --- ccx = dwconv(x) @ cc_pw  (N=384) ---
  k_gemm_bf16<<<dim3(DIM / 64, ROWS / 128), T, 0, stream>>>(
      Ybf, DMODEL, CcpwT, DMODEL, cc_b, nullptr, ccx, DIM, DMODEL);

  // --- head split + local branch ---
  k_split_heads<<<(BH * SEQ * HSIZE + T - 1) / T, T, 0, stream>>>(qkvc, Qh, Kh, Vt);
  k_ck<<<ROWS, 64, 0, stream>>>(qkvc, ccx, ck_w, ck_b, lw);
  k_local<<<(ROWS * DIM + T - 1) / T, T, 0, stream>>>(lw, qkvc, catbf);

  // --- global attention ---
  k_scores<<<dim3(SEQ / 64, SEQ / 128, BH), T, 0, stream>>>(Qh, Kh, mask, a1);
  k_softmax<<<BH * SEQ, T, 0, stream>>>(a1);
  k_ctx<<<dim3(1, SEQ / 128, BH), T, 0, stream>>>(a1, Vt, catbf);

  // --- dense + residual + LayerNorm ---
  k_gemm_bf16<<<dim3(DMODEL / 64, ROWS / 128), T, 0, stream>>>(
      catbf, 2 * DIM, DenseT, DMODEL, dense_b, x, hbuf, DMODEL, 2 * DIM);
  k_layernorm<<<ROWS, T, 0, stream>>>(hbuf, gamma, beta, out);
}
